// MultiHeadAttention_36507222016215
// MI455X (gfx1250) — matrix operand order repaired and checked by reading, not compiled, not run on hardware
//
#include <hip/hip_runtime.h>
#include <hip/hip_bf16.h>

// ---------------------------------------------------------------------------
// MI455X (gfx1250) multi-head attention with relative position bias.
// All GEMM-shaped work (projections, QK^T, bias einsum, P*V, output proj)
// runs on v_wmma_f32_16x16x32_bf16 with fp32 accumulation.
// The bias einsum 'bjhd,jkd->bhjk' is batched per-j as [128x64]@[64x1024]
// WMMA GEMMs into a 33.5MB bf16 tile that stays resident in the 192MB L2,
// interleaved with the attention kernel over 8 j-blocks.
// Attention K-tiles are staged with global_load_async_to_lds_b128 in a
// 2-stage double-buffered pipeline (DMA for tile i+1 overlaps compute on i).
// Workspace: 4 x 16.78MB (qw/kw/vw/o) + 33.55MB bias tile ~= 101 MB.
// ---------------------------------------------------------------------------

typedef __bf16 bf16;
typedef __attribute__((ext_vector_type(16))) __bf16 v16bf;
typedef __attribute__((ext_vector_type(8)))  __bf16 bf16x8;
typedef __attribute__((ext_vector_type(8)))  float  v8f;
typedef __attribute__((ext_vector_type(4)))  float  v4f;

constexpr int Bc = 8, Sc = 1024, Dc = 1024, Hc = 16, DKc = 64;
constexpr int HDc = Hc * DKc;        // 1024
constexpr int MROWS = Bc * Sc;       // 8192
constexpr int JBLK = 128;            // j rows per bias block (8 blocks)

__device__ inline bf16x8 ld8(const bf16* p) {
  return *reinterpret_cast<const bf16x8*>(p);
}

__device__ inline v16bf pack16(bf16x8 lo, bf16x8 hi) {
  v16bf r;
#pragma unroll
  for (int i = 0; i < 8; ++i) { r[i] = lo[i]; r[i + 8] = hi[i]; }
  return r;
}

__device__ inline v8f wmma_bf16(v16bf a, v16bf b, v8f c) {
  return __builtin_amdgcn_wmma_f32_16x16x32_bf16(false, a, false, b, (short)0,
                                                 c, false, false);
}

// CDNA5 async global->LDS copy (16B per lane), tracked by ASYNCcnt.
__device__ inline void async_load_b128(unsigned lds_off, const void* gptr) {
  asm volatile("global_load_async_to_lds_b128 %0, %1, off"
               :: "v"(lds_off), "v"(gptr) : "memory");
}
__device__ inline void wait_asynccnt0() {
  asm volatile("s_wait_asynccnt 0" ::: "memory");
}

// Load 16 contiguous elements (converting fp32 -> bf16 if needed) into t[0..15]
__device__ inline void load16(const float* __restrict__ g, bf16* t) {
#pragma unroll
  for (int i = 0; i < 16; i += 4) {
    v4f x = *reinterpret_cast<const v4f*>(g + i);
    t[i + 0] = (bf16)x[0]; t[i + 1] = (bf16)x[1];
    t[i + 2] = (bf16)x[2]; t[i + 3] = (bf16)x[3];
  }
}
__device__ inline void load16(const bf16* __restrict__ g, bf16* t) {
  *reinterpret_cast<bf16x8*>(t)     = ld8(g);
  *reinterpret_cast<bf16x8*>(t + 8) = ld8(g + 8);
}

// ---------------------------------------------------------------------------
// Tiled GEMM: C[M,N] = A[M,K] @ B[K,N] + bias[N], optional per-row int mask.
// Block tile 128x128, K-step 32, 8 waves (wave32), wave tile 64x32.
// 2-stage software pipeline: global loads for step i+1 overlap WMMAs of i;
// double-buffered LDS, one barrier per K-step.
// ---------------------------------------------------------------------------
template <typename AT, typename BT, bool OUT_BF16>
__global__ __launch_bounds__(256) void gemm_wmma_kernel(
    const AT* __restrict__ A, const BT* __restrict__ Bm,
    const float* __restrict__ bias, const int* __restrict__ rowmask,
    void* __restrict__ Cvoid, int M, int N, int K) {
  __shared__ bf16 As[2][128][32];
  __shared__ bf16 BsT[2][128][32];   // [n][k] (transposed during scatter)

  const int tid  = threadIdx.x;
  const int lane = tid & 31;
  const int wid  = tid >> 5;
  const int wm   = wid >> 2;      // 0..1
  const int wn   = wid & 3;       // 0..3
  const int m0   = blockIdx.y * 128;
  const int n0   = blockIdx.x * 128;

  const v8f vzero = {};
  v8f acc[4][2];
#pragma unroll
  for (int mt = 0; mt < 4; ++mt)
#pragma unroll
    for (int nt = 0; nt < 2; ++nt) acc[mt][nt] = vzero;

  const int ar = tid >> 1, ac = (tid & 1) * 16;   // A staging: 128 rows x 32
  const int bk = tid >> 3, bc = (tid & 7) * 16;   // B staging: 32 k x 128 n
  const int rA  = lane & 15;
  const int kbA = (lane < 16) ? 0 : 8;            // A-frag K split (ISA 7.12.2)
  const int kbB = (lane < 16) ? 0 : 8;            // B-frag K split

  // Pipeline registers: data for the K-step about to be scattered.
  alignas(16) bf16 atmp[16];
  alignas(16) bf16 btmp[16];
  load16(A + (size_t)(m0 + ar) * K + ac, atmp);
  load16(Bm + (size_t)bk * N + (n0 + bc), btmp);

  int buf = 0;
  for (int k0 = 0; k0 < K; k0 += 32) {
    // Scatter current K-step into LDS buffer 'buf'.
    *reinterpret_cast<bf16x8*>(&As[buf][ar][ac]) =
        *reinterpret_cast<bf16x8*>(atmp);
    *reinterpret_cast<bf16x8*>(&As[buf][ar][ac + 8]) =
        *reinterpret_cast<bf16x8*>(atmp + 8);
#pragma unroll
    for (int i = 0; i < 16; ++i) BsT[buf][bc + i][bk] = btmp[i];
    __syncthreads();

    // Issue global loads for the next K-step (overlap with WMMAs below).
    if (k0 + 32 < K) {
      load16(A + (size_t)(m0 + ar) * K + (k0 + 32 + ac), atmp);
      load16(Bm + (size_t)(k0 + 32 + bk) * N + (n0 + bc), btmp);
    }

    v16bf af[4], bfr[2];
#pragma unroll
    for (int mt = 0; mt < 4; ++mt) {
      const bf16* p = &As[buf][wm * 64 + mt * 16 + rA][0];
      af[mt] = pack16(ld8(p + kbA), ld8(p + kbA + 16));
    }
#pragma unroll
    for (int nt = 0; nt < 2; ++nt) {
      const bf16* p = &BsT[buf][wn * 32 + nt * 16 + rA][0];
      bfr[nt] = pack16(ld8(p + kbB), ld8(p + kbB + 16));
    }
#pragma unroll
    for (int mt = 0; mt < 4; ++mt)
#pragma unroll
      for (int nt = 0; nt < 2; ++nt)
        acc[mt][nt] = wmma_bf16(af[mt], bfr[nt], acc[mt][nt]);
    buf ^= 1;
  }

  const int rhalf = (lane >> 4) * 8;   // C/D layout: VGPR r -> M = r (+8 hi half)
#pragma unroll
  for (int mt = 0; mt < 4; ++mt)
#pragma unroll
    for (int nt = 0; nt < 2; ++nt)
#pragma unroll
      for (int r = 0; r < 8; ++r) {
        const int m = m0 + wm * 64 + mt * 16 + rhalf + r;
        const int n = n0 + wn * 32 + nt * 16 + (lane & 15);
        float vv = acc[mt][nt][r] + bias[n];
        if (rowmask) vv *= (float)rowmask[m];
        if (OUT_BF16)
          reinterpret_cast<bf16*>(Cvoid)[(size_t)m * N + n] = (bf16)vv;
        else
          reinterpret_cast<float*>(Cvoid)[(size_t)m * N + n] = vv;
      }
}

// ---------------------------------------------------------------------------
// Bias einsum as WMMA GEMM, batched over j.
// For fixed j:  bias[bh, k] = qw[bh(j)] [128 x 64] @ pos_bias[j]^T [64 x 1024].
// Grid (S/128 key-blocks, JBLK j-rows); K=64 fully staged in LDS.
// Output: bias_buf[jloc][bh][k] bf16 (33.5MB, L2-resident).
// ---------------------------------------------------------------------------
__global__ __launch_bounds__(256) void bias_gemm_kernel(
    const bf16* __restrict__ qw, const float* __restrict__ pos_bias,
    bf16* __restrict__ bias_buf, int jb) {
  __shared__ bf16 As[128][64];   // rows = bh = b*16+h
  __shared__ bf16 Bs[128][64];   // rows = key, cols = d

  const int tid  = threadIdx.x;
  const int lane = tid & 31;
  const int wid  = tid >> 5;
  const int wm   = wid >> 2;     // 0..1
  const int wn   = wid & 3;      // 0..3
  const int k0   = blockIdx.x * 128;
  const int jloc = blockIdx.y;
  const int j    = jb * JBLK + jloc;

  // Stage A (gather 128 strided qw rows) and B (fp32 -> bf16, no transpose:
  // Bs[key][d] already matches the B-frag layout: lane n = key, elems = d).
  {
    const int r = tid >> 1, c = (tid & 1) * 32;
    const int bb = r >> 4, hh = r & 15;
    const bf16* ga = qw + (size_t)(bb * Sc + j) * HDc + hh * DKc + c;
#pragma unroll
    for (int i = 0; i < 32; i += 8)
      *reinterpret_cast<bf16x8*>(&As[r][c + i]) = ld8(ga + i);
    const float* gb = pos_bias + ((size_t)j * Sc + (k0 + r)) * DKc + c;
#pragma unroll
    for (int i = 0; i < 32; i += 4) {
      v4f x = *reinterpret_cast<const v4f*>(gb + i);
      Bs[r][c + i + 0] = (bf16)x[0]; Bs[r][c + i + 1] = (bf16)x[1];
      Bs[r][c + i + 2] = (bf16)x[2]; Bs[r][c + i + 3] = (bf16)x[3];
    }
  }
  __syncthreads();

  const v8f vzero = {};
  v8f acc[4][2];
#pragma unroll
  for (int mt = 0; mt < 4; ++mt)
#pragma unroll
    for (int nt = 0; nt < 2; ++nt) acc[mt][nt] = vzero;

  const int rA  = lane & 15;
  const int kbA = (lane < 16) ? 0 : 8;
  const int kbB = (lane < 16) ? 0 : 8;
#pragma unroll
  for (int d0 = 0; d0 < 64; d0 += 32) {
    v16bf af[4], bfr[2];
#pragma unroll
    for (int mt = 0; mt < 4; ++mt) {
      const bf16* p = &As[wm * 64 + mt * 16 + rA][d0];
      af[mt] = pack16(ld8(p + kbA), ld8(p + kbA + 16));
    }
#pragma unroll
    for (int nt = 0; nt < 2; ++nt) {
      const bf16* p = &Bs[wn * 32 + nt * 16 + rA][d0];
      bfr[nt] = pack16(ld8(p + kbB), ld8(p + kbB + 16));
    }
#pragma unroll
    for (int mt = 0; mt < 4; ++mt)
#pragma unroll
      for (int nt = 0; nt < 2; ++nt)
        acc[mt][nt] = wmma_bf16(af[mt], bfr[nt], acc[mt][nt]);
  }

  const int rhalf = (lane >> 4) * 8;
#pragma unroll
  for (int mt = 0; mt < 4; ++mt)
#pragma unroll
    for (int nt = 0; nt < 2; ++nt)
#pragma unroll
      for (int r = 0; r < 8; ++r) {
        const int bh = wm * 64 + mt * 16 + rhalf + r;
        const int k  = k0 + wn * 32 + nt * 16 + (lane & 15);
        bias_buf[((size_t)jloc * (Bc * Hc) + bh) * Sc + k] =
            (bf16)acc[mt][nt][r];
      }
}

// ---------------------------------------------------------------------------
// Fused flash attention over one 128-row j-block.
// Grid (JBLK/64, H, B); 128 threads = 4 waves; wave owns 16 query rows.
// scores = (QK^T [WMMA] + bias_buf [precomputed via WMMA]) / 8 - (1-vmask)*NEG
// 2-stage pipeline: K tile async-DMA'd (ASYNCcnt) and V tile loaded to regs
// for step i+1 while step i computes; double-buffered LDS, 1 barrier/step.
// ---------------------------------------------------------------------------
__global__ __launch_bounds__(128) void attn_kernel(
    const bf16* __restrict__ qw, const bf16* __restrict__ kwm,
    const bf16* __restrict__ vwm, const bf16* __restrict__ bias_buf,
    const int* __restrict__ v_mask, bf16* __restrict__ o_out, int jb) {
  __shared__ bf16 kws[2][32][64];   // K tile, row-major [k][d] (async-staged)
  __shared__ bf16 vwsT[2][64][32];  // V tile transposed [d][k]
  __shared__ bf16 pbuf[4][16][32];  // per-wave P staging (C-layout -> A-frag)

  const int tid  = threadIdx.x;
  const int lane = tid & 31;
  const int w    = tid >> 5;
  const int jt   = blockIdx.x;               // 64-row tile within j-block
  const int h    = blockIdx.y;
  const int b    = blockIdx.z;
  const int bh   = b * Hc + h;
  const int j0   = jb * JBLK + jt * 64 + w * 16;
  const int rA   = lane & 15;
  const int rhalf = (lane >> 4) * 8;
  const int kbA  = (lane < 16) ? 0 : 8;
  const int kbB  = (lane < 16) ? 0 : 8;

  // Q A-fragments (d = 0..31 and 32..63), kept in registers all kernel.
  v16bf aq[2];
  {
    const bf16* qrow = qw + (size_t)(b * Sc + j0 + rA) * HDc + h * DKc;
#pragma unroll
    for (int t = 0; t < 2; ++t)
      aq[t] = pack16(ld8(qrow + t * 32 + kbA), ld8(qrow + t * 32 + kbA + 16));
  }

  float rowmax[8], rowsum[8];
  const v8f vzero = {};
  v8f accO[4];
#pragma unroll
  for (int r = 0; r < 8; ++r) { rowmax[r] = -3.0e38f; rowsum[r] = 0.0f; }
#pragma unroll
  for (int dt = 0; dt < 4; ++dt) accO[dt] = vzero;

  // Pipeline prologue: async K(0) into buffer 0, V(0) into registers.
  const int kr = tid >> 2, c = (tid & 3) * 16;   // staging: 32 keys x 64 dims
  alignas(16) bf16 vcur[16];
  {
    const bf16* gk = kwm + (size_t)(b * Sc + kr) * HDc + h * DKc + c;
    async_load_b128((unsigned)(size_t)&kws[0][kr][c], gk);
    const bf16* gv = vwm + (size_t)(b * Sc + kr) * HDc + h * DKc + c;
    *reinterpret_cast<bf16x8*>(vcur)     = ld8(gv);
    *reinterpret_cast<bf16x8*>(vcur + 8) = ld8(gv + 8);
  }

  int buf = 0;
  for (int kt = 0; kt < Sc; kt += 32) {
    // Scatter V(kt) from pipeline regs into vwsT[buf] (disjoint from the
    // buffer any lagging wave may still be reading -> safe pre-barrier).
#pragma unroll
    for (int i = 0; i < 8; ++i) {
      vwsT[buf][c + i][kr]     = vcur[i];
      vwsT[buf][c + 8 + i][kr] = vcur[8 + i];
    }
    // Own async K(kt) must land before the barrier for cross-wave visibility.
    wait_asynccnt0();
    __syncthreads();

    // Kick off next tile's K DMA + V loads; they fly during compute below.
    if (kt + 32 < Sc) {
      const bf16* gk =
          kwm + (size_t)(b * Sc + kt + 32 + kr) * HDc + h * DKc + c;
      async_load_b128((unsigned)(size_t)&kws[buf ^ 1][kr][c], gk);
      const bf16* gv =
          vwm + (size_t)(b * Sc + kt + 32 + kr) * HDc + h * DKc + c;
      *reinterpret_cast<bf16x8*>(vcur)     = ld8(gv);
      *reinterpret_cast<bf16x8*>(vcur + 8) = ld8(gv + 8);
      // Prefetch next k-tile of the L2-resident bias rows for this wave.
      __builtin_prefetch(
          &bias_buf[((size_t)(jt * 64 + w * 16 + rA) * (Bc * Hc) + bh) * Sc +
                    kt + 32], 0, 1);
    }

    float st[2][8];
#pragma unroll
    for (int nt = 0; nt < 2; ++nt) {
      // QK^T 16x16 tile via two chained WMMAs over d.
      v8f sac = vzero;
      const int kcol = nt * 16 + rA;
#pragma unroll
      for (int t = 0; t < 2; ++t) {
        const bf16* p = &kws[buf][kcol][t * 32 + kbB];
        v16bf bk = pack16(ld8(p), ld8(p + 16));
        sac = wmma_bf16(aq[t], bk, sac);
      }
      // + precomputed relative-position bias, scale, key mask.
      const int kg = kt + nt * 16 + rA;
      const float vmv = (float)v_mask[b * Sc + kg];
#pragma unroll
      for (int r = 0; r < 8; ++r) {
        const int ml = r + rhalf;
        const int jloc = jt * 64 + w * 16 + ml;
        const float bias =
            (float)bias_buf[((size_t)jloc * (Bc * Hc) + bh) * Sc + kg];
        float s = (sac[r] + bias) * 0.125f;   // 1/sqrt(DK)
        s -= (1.0f - vmv) * 1.0e12f;
        st[nt][r] = s;
      }
    }

    // Online softmax (row lives across 16 lanes of a half-wave).
#pragma unroll
    for (int r = 0; r < 8; ++r) {
      float tmax = fmaxf(st[0][r], st[1][r]);
#pragma unroll
      for (int o = 8; o >= 1; o >>= 1) tmax = fmaxf(tmax, __shfl_xor(tmax, o, 32));
      const float mnew = fmaxf(rowmax[r], tmax);
      const float corr = __expf(rowmax[r] - mnew);
      rowmax[r] = mnew;
      const float p0 = __expf(st[0][r] - mnew);
      const float p1 = __expf(st[1][r] - mnew);
      float ts = p0 + p1;
#pragma unroll
      for (int o = 8; o >= 1; o >>= 1) ts += __shfl_xor(ts, o, 32);
      rowsum[r] = rowsum[r] * corr + ts;
#pragma unroll
      for (int dt = 0; dt < 4; ++dt) accO[dt][r] *= corr;
      const int ml = r + rhalf;
      pbuf[w][ml][rA]      = (bf16)p0;
      pbuf[w][ml][16 + rA] = (bf16)p1;
    }

    // P (C-layout) -> A-frag through per-wave LDS buffer, then O += P @ V.
    v16bf ap;
    {
      const bf16* p = &pbuf[w][rA][0];
      ap = pack16(ld8(p + kbA), ld8(p + kbA + 16));
    }
#pragma unroll
    for (int dt = 0; dt < 4; ++dt) {
      const int dcol = dt * 16 + rA;
      const bf16* p = &vwsT[buf][dcol][kbB];
      v16bf bv = pack16(ld8(p), ld8(p + 16));
      accO[dt] = wmma_bf16(ap, bv, accO[dt]);
    }
    buf ^= 1;
  }

  // Normalize and emit bf16 attention output [B,S,H*DK].
#pragma unroll
  for (int r = 0; r < 8; ++r) {
    const int ml = r + rhalf;
    const float inv = 1.0f / rowsum[r];
    const int jg = jb * JBLK + jt * 64 + w * 16 + ml;
#pragma unroll
    for (int dt = 0; dt < 4; ++dt) {
      const int d = dt * 16 + (lane & 15);
      o_out[(size_t)(b * Sc + jg) * HDc + h * DKc + d] = (bf16)(accO[dt][r] * inv);
    }
  }
}

// ---------------------------------------------------------------------------
extern "C" void kernel_launch(void* const* d_in, const int* in_sizes, int n_in,
                              void* d_out, int out_size, void* d_ws,
                              size_t ws_size, hipStream_t stream) {
  (void)in_sizes; (void)n_in; (void)out_size; (void)ws_size;
  const float* q        = (const float*)d_in[0];
  const float* k_unused = (const float*)d_in[1];  (void)k_unused; // quirk: K from v
  const float* v        = (const float*)d_in[2];
  const float* pos_bias = (const float*)d_in[3];
  const float* Wq       = (const float*)d_in[4];
  const float* Wk       = (const float*)d_in[5];
  const float* Wv       = (const float*)d_in[6];
  const float* Wo       = (const float*)d_in[7];
  const float* bq       = (const float*)d_in[8];
  const float* bk       = (const float*)d_in[9];
  const float* bv       = (const float*)d_in[10];
  const float* bo       = (const float*)d_in[11];
  const int*   q_mask   = (const int*)d_in[12];
  const int*   v_mask   = (const int*)d_in[13];

  char* ws = (char*)d_ws;
  size_t off = 0;
  auto take = [&](size_t bytes) -> void* {
    void* p = ws + off;
    off += (bytes + 255) & ~(size_t)255;
    return p;
  };
  bf16* qw = (bf16*)take((size_t)MROWS * HDc * sizeof(bf16));
  bf16* kw = (bf16*)take((size_t)MROWS * HDc * sizeof(bf16));
  bf16* vw = (bf16*)take((size_t)MROWS * HDc * sizeof(bf16));
  bf16* ob = (bf16*)take((size_t)MROWS * HDc * sizeof(bf16));
  bf16* bias_buf = (bf16*)take((size_t)JBLK * (Bc * Hc) * Sc * sizeof(bf16));

  const dim3 gblk(256);
  const dim3 ggrid(HDc / 128, MROWS / 128);

  // Projections (fp32 inputs converted to bf16 during LDS staging).
  gemm_wmma_kernel<float, float, true>
      <<<ggrid, gblk, 0, stream>>>(q, Wq, bq, nullptr, qw, MROWS, HDc, Dc);
  gemm_wmma_kernel<float, float, true>
      <<<ggrid, gblk, 0, stream>>>(v, Wk, bk, nullptr, kw, MROWS, HDc, Dc);
  gemm_wmma_kernel<float, float, true>
      <<<ggrid, gblk, 0, stream>>>(v, Wv, bv, nullptr, vw, MROWS, HDc, Dc);

  // Interleave per-j-block bias WMMA GEMM (L2-resident tile) with attention.
  for (int jb = 0; jb < Sc / JBLK; ++jb) {
    bias_gemm_kernel<<<dim3(Sc / 128, JBLK), gblk, 0, stream>>>(
        qw, pos_bias, bias_buf, jb);
    attn_kernel<<<dim3(JBLK / 64, Hc, Bc), dim3(128), 0, stream>>>(
        qw, kw, vw, bias_buf, v_mask, ob, jb);
  }

  // Output projection + bias + multiplicative query mask -> fp32 d_out.
  gemm_wmma_kernel<bf16, float, false>
      <<<ggrid, gblk, 0, stream>>>(ob, Wo, bo, q_mask, d_out, MROWS, Dc, HDc);
}
